// TransformerBlock_25434796327439
// MI455X (gfx1250) — matrix operand order repaired and checked by reading, not compiled, not run on hardware
//
#include <hip/hip_runtime.h>
#include <hip/hip_bf16.h>
#include <math.h>

// ---------------------------------------------------------------------------
// ViT transformer block (B=64, T=197, D=768, H=12, Dh=64, HID=3072) for
// gfx1250 (MI455X). All GEMMs + both attention einsums run on
// v_wmma_f32_16x16x32_f16 (f16 in, f32 accumulate). Attention is fused
// flash-style per (batch, head); the K tile is staged to LDS by the Tensor
// Data Mover (tensor_load_to_lds + s_wait_tensorcnt). ws_size >= ~150 MB.
// ---------------------------------------------------------------------------

typedef __attribute__((ext_vector_type(16))) _Float16 v16h;
typedef __attribute__((ext_vector_type(8)))  _Float16 v8h;
typedef __attribute__((ext_vector_type(8)))  float    v8f;
typedef __attribute__((ext_vector_type(4)))  unsigned int u32x4;
typedef __attribute__((ext_vector_type(8)))  int          i32x8;
typedef __attribute__((ext_vector_type(4)))  int          i32x4;

#define WMMA_F16(A, B, C) \
  __builtin_amdgcn_wmma_f32_16x16x32_f16(false, (A), false, (B), (short)0, (C), false, false)

#if __has_builtin(__builtin_amdgcn_tensor_load_to_lds) && \
    __has_builtin(__builtin_amdgcn_s_wait_tensorcnt)
#define HAVE_TDM 1
#else
#define HAVE_TDM 0
#endif

#if HAVE_TDM
// 2D TDM: copy tile_h x tile_w (f16 elements, row stride `stride` elements)
// from global into LDS at byte offset lds_addr. Rows >= tensor_h read as 0
// (TDM OOB rule) -> free zero padding. Issue from one wave; drain TENSORcnt.
__device__ __forceinline__ void tdm_load_2d_f16(
    unsigned lds_addr, const void* gptr, unsigned tile_w, unsigned tile_h,
    unsigned tensor_w, unsigned tensor_h, unsigned stride) {
  const unsigned long long ga = (unsigned long long)(size_t)gptr;
  u32x4 g0;
  g0[0] = 1u;                                   // count=1, user descriptor
  g0[1] = lds_addr;                             // LDS byte address
  g0[2] = (unsigned)(ga & 0xFFFFFFFFu);         // global_addr[31:0]
  g0[3] = (unsigned)((ga >> 32) & 0x1FFFFFFu)   // global_addr[56:32]
          | (2u << 30);                         // type = 2 ("image")
  i32x8 g1;
  g1[0] = (int)(1u << 16);                      // data_size = 1 -> 2 bytes
  g1[1] = (int)(tensor_w << 16);                // tensor_dim0[15:0]  @ bit 48
  g1[2] = (int)(tensor_h << 16);                // tensor_dim1[15:0]  @ bit 80
  g1[3] = (int)(tile_w << 16);                  // tile_dim0          @ bit 112
  g1[4] = (int)tile_h;                          // tile_dim1          @ bit 128
  g1[5] = (int)stride;                          // tensor_dim0_stride @ bit 160
  g1[6] = 0;
  g1[7] = 0;
  const i32x4 z4 = {0, 0, 0, 0};
#if defined(__clang_major__) && __clang_major__ >= 23
  const i32x8 z8 = {0, 0, 0, 0, 0, 0, 0, 0};
  __builtin_amdgcn_tensor_load_to_lds(g0, g1, z4, z4, z8, 0);
#else
  __builtin_amdgcn_tensor_load_to_lds(g0, g1, z4, z4, 0);
#endif
}
#endif

// A-fragment (16x32): lanes 0-15 -> row=lane, K={0..15};
// lanes 16-31 -> row=lane-16, K={16..31}.
__device__ __forceinline__ v16h load_frag_a(const _Float16* p) {
  v8h lo = *(const v8h*)(p);
  v8h hi = *(const v8h*)(p + 8);
  v16h a;
#pragma unroll
  for (int i = 0; i < 8; ++i) { a[i] = lo[i]; a[i + 8] = hi[i]; }
  return a;
}

// ---------------------------------------------------------------------------
// Weight transpose + f32->f16 convert: Wt[n*K + k] = (f16) W[k*N + n]
// ---------------------------------------------------------------------------
__global__ __launch_bounds__(256) void transpose_to_f16(
    const float* __restrict__ W, _Float16* __restrict__ Wt, int K, int N) {
  __shared__ float tile[32][33];
  const int n0 = blockIdx.x * 32, k0 = blockIdx.y * 32;
  const int tx = threadIdx.x & 31, ty = threadIdx.x >> 5;  // 32 x 8
#pragma unroll
  for (int i = 0; i < 32; i += 8)
    tile[ty + i][tx] = W[(size_t)(k0 + ty + i) * N + (n0 + tx)];
  __syncthreads();
#pragma unroll
  for (int i = 0; i < 32; i += 8)
    Wt[(size_t)(n0 + ty + i) * K + (k0 + tx)] = (_Float16)tile[tx][ty + i];
}

// ---------------------------------------------------------------------------
// LayerNorm over D=768, output f16. One 256-thread block per token.
// ---------------------------------------------------------------------------
__global__ __launch_bounds__(256) void layernorm_to_f16(
    const float* __restrict__ x, const float* __restrict__ g,
    const float* __restrict__ bb, _Float16* __restrict__ out) {
  const int t = blockIdx.x;
  const float* xr = x + (size_t)t * 768;
  const int i0 = threadIdx.x, i1 = i0 + 256, i2 = i0 + 512;
  const float v0 = xr[i0], v1 = xr[i1], v2 = xr[i2];
  float s = v0 + v1 + v2;
  float q = v0 * v0 + v1 * v1 + v2 * v2;
#pragma unroll
  for (int m = 1; m < 32; m <<= 1) {
    s += __shfl_xor(s, m, 32);
    q += __shfl_xor(q, m, 32);
  }
  __shared__ float rs[8], rq[8];
  if ((threadIdx.x & 31) == 0) { rs[threadIdx.x >> 5] = s; rq[threadIdx.x >> 5] = q; }
  __syncthreads();
  s = 0.0f; q = 0.0f;
#pragma unroll
  for (int w = 0; w < 8; ++w) { s += rs[w]; q += rq[w]; }
  const float mean = s * (1.0f / 768.0f);
  const float var  = q * (1.0f / 768.0f) - mean * mean;
  const float rstd = rsqrtf(var + 1e-5f);
  _Float16* o = out + (size_t)t * 768;
  o[i0] = (_Float16)((v0 - mean) * rstd * g[i0] + bb[i0]);
  o[i1] = (_Float16)((v1 - mean) * rstd * g[i1] + bb[i1]);
  o[i2] = (_Float16)((v2 - mean) * rstd * g[i2] + bb[i2]);
}

// ---------------------------------------------------------------------------
// WMMA GEMM: C(MxN) = A(MxK f16 row-major) x Bt(NxK f16 row-major = B^T)
// MODE 0: outH = (f16) acc                       (QKV projections)
// MODE 1: outF = resid + acc + bias  (f32)       (Wo/W2 + residual)
// MODE 2: outH = (f16) gelu_exact(acc + bias)    (W1 + GELU)
// Wave tile 32M x 64N (2 A-frags reuse 4 B-frags -> 8 WMMA per K-step);
// block = 8 waves = 64M x 256N. M%64==0, N%256==0, K%32==0 all hold.
// ---------------------------------------------------------------------------
template <int MODE>
__global__ __launch_bounds__(256) void gemm_wmma(
    const _Float16* __restrict__ A, const _Float16* __restrict__ Bt,
    const float* __restrict__ bias, const float* __restrict__ resid,
    float* __restrict__ outF, _Float16* __restrict__ outH,
    int M, int N, int K) {
  const int lane = threadIdx.x & 31;
  const int wave = threadIdx.x >> 5;
  const int row0 = blockIdx.y * 64 + (wave >> 2) * 32;
  const int col0 = blockIdx.x * 256 + (wave & 3) * 64;

  const int arow = lane & 15;
  const int akb  = (lane >> 4) * 16;
  const int bcol = lane & 15;
  const int bkb  = (lane >> 4) * 16;

  const _Float16* arp0 = A  + (size_t)(row0 + arow) * K + akb;
  const _Float16* arp1 = arp0 + (size_t)16 * K;
  const _Float16* bp0 = Bt + (size_t)(col0 +  0 + bcol) * K + bkb;
  const _Float16* bp1 = Bt + (size_t)(col0 + 16 + bcol) * K + bkb;
  const _Float16* bp2 = Bt + (size_t)(col0 + 32 + bcol) * K + bkb;
  const _Float16* bp3 = Bt + (size_t)(col0 + 48 + bcol) * K + bkb;

  v8f a00 = {}, a01 = {}, a02 = {}, a03 = {};
  v8f a10 = {}, a11 = {}, a12 = {}, a13 = {};

  for (int k = 0; k < K; k += 32) {
    if (k + 128 < K) {
      __builtin_prefetch(arp0 + k + 128, 0, 3);
      __builtin_prefetch(arp1 + k + 128, 0, 3);
      __builtin_prefetch(bp0 + k + 128, 0, 3);
      __builtin_prefetch(bp2 + k + 128, 0, 3);
    }
    const v16h fa0 = load_frag_a(arp0 + k);
    const v16h fa1 = load_frag_a(arp1 + k);
    const v16h b0 = *(const v16h*)(bp0 + k);
    const v16h b1 = *(const v16h*)(bp1 + k);
    const v16h b2 = *(const v16h*)(bp2 + k);
    const v16h b3 = *(const v16h*)(bp3 + k);
    a00 = WMMA_F16(fa0, b0, a00);
    a01 = WMMA_F16(fa0, b1, a01);
    a02 = WMMA_F16(fa0, b2, a02);
    a03 = WMMA_F16(fa0, b3, a03);
    a10 = WMMA_F16(fa1, b0, a10);
    a11 = WMMA_F16(fa1, b1, a11);
    a12 = WMMA_F16(fa1, b2, a12);
    a13 = WMMA_F16(fa1, b3, a13);
  }

#pragma unroll
  for (int mh = 0; mh < 2; ++mh) {
#pragma unroll
    for (int r = 0; r < 8; ++r) {
      const int row = row0 + mh * 16 + r + (lane >> 4) * 8;
      const int cb  = col0 + (lane & 15);
      const float vals[4] = {
          mh ? a10[r] : a00[r], mh ? a11[r] : a01[r],
          mh ? a12[r] : a02[r], mh ? a13[r] : a03[r]};
#pragma unroll
      for (int j = 0; j < 4; ++j) {
        const int col = cb + j * 16;
        const size_t idx = (size_t)row * N + col;
        const float v = vals[j];
        if (MODE == 0) {
          outH[idx] = (_Float16)v;
        } else if (MODE == 1) {
          outF[idx] = resid[idx] + v + bias[col];
        } else {
          const float tt = v + bias[col];
          outH[idx] = (_Float16)(0.5f * tt * (1.0f + erff(tt * 0.70710678118654752f)));
        }
      }
    }
  }
}

// ---------------------------------------------------------------------------
// Fused flash attention, one block per (batch, head). q/k/v f16 in
// (b, t, h*64+d) layout. K tile staged to LDS by the TDM (zero-padded keys
// via the tensor OOB rule); V^T staged manually; scores and P*V on WMMA.
// Static LDS = 28672 + 28672 + 8192 = exactly 64 KB.
// ---------------------------------------------------------------------------
__global__ __launch_bounds__(256) void attn_fused(
    const _Float16* __restrict__ Q, const _Float16* __restrict__ Km,
    const _Float16* __restrict__ V, _Float16* __restrict__ ctx) {
  __shared__ alignas(128) _Float16 Kl[224][64];    // [key][feat], TDM target
  __shared__ alignas(32)  _Float16 Vt[64][224];    // [feat][key]
  __shared__ alignas(32)  _Float16 Pb[8][16][32];  // per-wave P transpose

  const int b = blockIdx.x / 12, h = blockIdx.x % 12;
  const size_t base = (size_t)b * 197 * 768 + (size_t)h * 64;

  const int lane = threadIdx.x & 31;
  const int wave = threadIdx.x >> 5;

#if HAVE_TDM
  if (wave == 0) {
    tdm_load_2d_f16((unsigned)(size_t)(&Kl[0][0]), (const void*)(Km + base),
                    /*tile_w=*/64, /*tile_h=*/224,
                    /*tensor_w=*/64, /*tensor_h=*/197, /*stride=*/768);
  }
#else
  for (int idx = threadIdx.x; idx < 224 * 64; idx += 256) {
    const int t = idx >> 6, d = idx & 63;
    Kl[t][d] = (t < 197) ? Km[base + (size_t)t * 768 + d] : (_Float16)0.0f;
  }
#endif

  // Stage V transposed into LDS (zero-pad keys 197..223).
  for (int idx = threadIdx.x; idx < 64 * 224; idx += 256) {
    const int t = idx / 64, d = idx % 64;
    Vt[d][t] = (t < 197) ? V[base + (size_t)t * 768 + d] : (_Float16)0.0f;
  }
#if HAVE_TDM
  if (wave == 0) __builtin_amdgcn_s_wait_tensorcnt(0);
#endif
  __syncthreads();

  const int arow = lane & 15;
  const int akb  = (lane >> 4) * 16;
  const int bcol = lane & 15;
  const int bkb  = (lane >> 4) * 16;

  for (int qi = 0; qi < 2; ++qi) {
    const int qt = wave + qi * 8;           // 13 query tiles over 8 waves
    if (qt > 12) break;                     // wave-uniform
    const int q0 = qt * 16;

    int qrow = q0 + arow; if (qrow > 196) qrow = 196;   // clamp tail rows
    const _Float16* qp = Q + base + (size_t)qrow * 768 + akb;
    const v16h aq0 = load_frag_a(qp);       // features 0..31
    const v16h aq1 = load_frag_a(qp + 32);  // features 32..63

    float mrow[8], lrow[8];
#pragma unroll
    for (int r = 0; r < 8; ++r) { mrow[r] = -1e30f; lrow[r] = 0.0f; }
    v8f o0 = {}, o1 = {}, o2 = {}, o3 = {};

    for (int kt = 0; kt < 224; kt += 32) {
      // ---- S tile: Q (16x64) x K^T (64x32) from LDS ----
      const _Float16* kp0 = &Kl[kt + bcol][bkb];
      const _Float16* kp1 = &Kl[kt + 16 + bcol][bkb];
      v8f s0 = {}, s1 = {};
      s0 = WMMA_F16(aq0, *(const v16h*)(kp0),      s0);
      s0 = WMMA_F16(aq1, *(const v16h*)(kp0 + 32), s0);
      s1 = WMMA_F16(aq0, *(const v16h*)(kp1),      s1);
      s1 = WMMA_F16(aq1, *(const v16h*)(kp1 + 32), s1);

      const int c0 = kt + bcol, c1 = c0 + 16;
#pragma unroll
      for (int r = 0; r < 8; ++r) {
        s0[r] = (c0 < 197) ? s0[r] * 0.125f : -1e30f;   // 1/sqrt(64) + pad mask
        s1[r] = (c1 < 197) ? s1[r] * 0.125f : -1e30f;
      }

      // ---- online softmax (row stats live per accumulator index r) ----
#pragma unroll
      for (int r = 0; r < 8; ++r) {
        float mx = fmaxf(s0[r], s1[r]);
#pragma unroll
        for (int msk = 1; msk < 16; msk <<= 1) mx = fmaxf(mx, __shfl_xor(mx, msk, 32));
        const float mn   = fmaxf(mrow[r], mx);
        const float corr = __expf(mrow[r] - mn);
        const float p0 = __expf(s0[r] - mn);
        const float p1 = __expf(s1[r] - mn);
        float ps = p0 + p1;
#pragma unroll
        for (int msk = 1; msk < 16; msk <<= 1) ps += __shfl_xor(ps, msk, 32);
        lrow[r] = lrow[r] * corr + ps;
        mrow[r] = mn;
        o0[r] *= corr; o1[r] *= corr; o2[r] *= corr; o3[r] *= corr;
        const int prow = r + (lane >> 4) * 8;
        Pb[wave][prow][bcol]      = (_Float16)p0;   // C-layout -> LDS
        Pb[wave][prow][bcol + 16] = (_Float16)p1;
      }
      // cross-lane LDS store -> load inside one wave: drain DS counter
      asm volatile("s_wait_dscnt 0" ::: "memory");

      // reread P as an A-fragment (16 q-rows x 32 keys)
      const v16h pa = load_frag_a(&Pb[wave][arow][akb]);

      // ---- O += P (16x32) x V (32x64), B-fragments from LDS V^T ----
      const _Float16* vb = &Vt[bcol][kt + bkb];
      o0 = WMMA_F16(pa, *(const v16h*)(vb),            o0);
      o1 = WMMA_F16(pa, *(const v16h*)(vb + 16 * 224), o1);
      o2 = WMMA_F16(pa, *(const v16h*)(vb + 32 * 224), o2);
      o3 = WMMA_F16(pa, *(const v16h*)(vb + 48 * 224), o3);
    }

#pragma unroll
    for (int r = 0; r < 8; ++r) {
      const int q = q0 + r + (lane >> 4) * 8;
      if (q < 197) {
        const float inv = 1.0f / lrow[r];
        const size_t o = base + (size_t)q * 768 + bcol;
        ctx[o]      = (_Float16)(o0[r] * inv);
        ctx[o + 16] = (_Float16)(o1[r] * inv);
        ctx[o + 32] = (_Float16)(o2[r] * inv);
        ctx[o + 48] = (_Float16)(o3[r] * inv);
      }
    }
  }
}

// ---------------------------------------------------------------------------
extern "C" void kernel_launch(void* const* d_in, const int* in_sizes, int n_in,
                              void* d_out, int out_size, void* d_ws, size_t ws_size,
                              hipStream_t stream) {
  (void)in_sizes; (void)n_in; (void)out_size; (void)ws_size;
  const float* x    = (const float*)d_in[0];
  const float* ln1g = (const float*)d_in[1];
  const float* ln1b = (const float*)d_in[2];
  const float* Wq   = (const float*)d_in[3];
  const float* Wk   = (const float*)d_in[4];
  const float* Wv   = (const float*)d_in[5];
  const float* Wo   = (const float*)d_in[6];
  const float* bo   = (const float*)d_in[7];
  const float* ln2g = (const float*)d_in[8];
  const float* ln2b = (const float*)d_in[9];
  const float* W1   = (const float*)d_in[10];
  const float* b1   = (const float*)d_in[11];
  const float* W2   = (const float*)d_in[12];
  const float* b2   = (const float*)d_in[13];
  float* out = (float*)d_out;
  char*  ws  = (char*)d_ws;

  constexpr int M = 64 * 197;   // 12608 (divisible by 64)
  constexpr int D = 768;
  constexpr int HID = 3072;

  size_t off = 0;
  auto alloc = [&](size_t bytes) {
    size_t o = off;
    off = (off + bytes + 255) & ~(size_t)255;
    return o;
  };
  _Float16* wq_t = (_Float16*)(ws + alloc((size_t)D * D * 2));
  _Float16* wk_t = (_Float16*)(ws + alloc((size_t)D * D * 2));
  _Float16* wv_t = (_Float16*)(ws + alloc((size_t)D * D * 2));
  _Float16* wo_t = (_Float16*)(ws + alloc((size_t)D * D * 2));
  _Float16* w1_t = (_Float16*)(ws + alloc((size_t)D * HID * 2));
  _Float16* w2_t = (_Float16*)(ws + alloc((size_t)HID * D * 2));
  _Float16* hbuf = (_Float16*)(ws + alloc((size_t)M * D * 2));   // h1, later h2
  _Float16* qbuf = (_Float16*)(ws + alloc((size_t)M * D * 2));
  _Float16* kbuf = (_Float16*)(ws + alloc((size_t)M * D * 2));
  _Float16* vbuf = (_Float16*)(ws + alloc((size_t)M * D * 2));
  _Float16* cbuf = (_Float16*)(ws + alloc((size_t)M * D * 2));
  float*    x2   = (float*)(ws + alloc((size_t)M * D * 4));
  _Float16* hid  = qbuf;  // q/k/v/ctx region (77.5 MB) reused for GELU out (77.4 MB)

  // 1) Weights -> f16, transposed to (N, K).
  transpose_to_f16<<<dim3(D / 32,   D / 32),   256, 0, stream>>>(Wq, wq_t, D, D);
  transpose_to_f16<<<dim3(D / 32,   D / 32),   256, 0, stream>>>(Wk, wk_t, D, D);
  transpose_to_f16<<<dim3(D / 32,   D / 32),   256, 0, stream>>>(Wv, wv_t, D, D);
  transpose_to_f16<<<dim3(D / 32,   D / 32),   256, 0, stream>>>(Wo, wo_t, D, D);
  transpose_to_f16<<<dim3(HID / 32, D / 32),   256, 0, stream>>>(W1, w1_t, D, HID);
  transpose_to_f16<<<dim3(D / 32,   HID / 32), 256, 0, stream>>>(W2, w2_t, HID, D);

  // 2) LN1 -> h (f16)
  layernorm_to_f16<<<M, 256, 0, stream>>>(x, ln1g, ln1b, hbuf);

  // 3) Q/K/V projections (WMMA)
  gemm_wmma<0><<<dim3(D / 256, M / 64), 256, 0, stream>>>(hbuf, wq_t, nullptr, nullptr, nullptr, qbuf, M, D, D);
  gemm_wmma<0><<<dim3(D / 256, M / 64), 256, 0, stream>>>(hbuf, wk_t, nullptr, nullptr, nullptr, kbuf, M, D, D);
  gemm_wmma<0><<<dim3(D / 256, M / 64), 256, 0, stream>>>(hbuf, wv_t, nullptr, nullptr, nullptr, vbuf, M, D, D);

  // 4) Fused flash attention per (batch, head)
  attn_fused<<<64 * 12, 256, 0, stream>>>(qbuf, kbuf, vbuf, cbuf);

  // 5) x2 = x + ctx @ Wo + bo   (f32)
  gemm_wmma<1><<<dim3(D / 256, M / 64), 256, 0, stream>>>(cbuf, wo_t, bo, x, x2, nullptr, M, D, D);

  // 6) LN2 -> h (f16)
  layernorm_to_f16<<<M, 256, 0, stream>>>(x2, ln2g, ln2b, hbuf);

  // 7) hid = gelu(h @ W1 + b1)  (f16)
  gemm_wmma<2><<<dim3(HID / 256, M / 64), 256, 0, stream>>>(hbuf, w1_t, b1, nullptr, nullptr, hid, M, HID, D);

  // 8) out = x2 + hid @ W2 + b2 (f32, final)
  gemm_wmma<1><<<dim3(D / 256, M / 64), 256, 0, stream>>>(hid, w2_t, b2, x2, out, nullptr, M, D, HID);
}
